// AttBlock_17068200034832
// MI455X (gfx1250) — matrix operand order repaired and checked by reading, not compiled, not run on hardware
//
#include <hip/hip_runtime.h>
#include <hip/hip_bf16.h>

typedef _Float16 half_t;
typedef __attribute__((ext_vector_type(16))) _Float16 v16h;
typedef __attribute__((ext_vector_type(8)))  float    v8f;

#define B_     4
#define C_CH   64
#define HW_DIM 128
#define P_     1024   // 32*32 window
#define NW_    64     // B * 4 * 4 windows

// ---------------- CDNA5 WMMA 16x16x32 f16 fragment loaders ----------------
// A: 16(M) x 32(K). lane: m = lane&15, g = (lane<16)?0:16
//    halves 0..15 -> k = k0 + g + h
__device__ inline v16h load_A_f16(const half_t* base, int ld, int lane, int k0) {
  const int m = lane & 15;
  const int g = (lane < 16) ? 0 : 16;
  const half_t* p = base + (size_t)m * ld + k0 + g;
  v16h a;
#pragma unroll
  for (int h = 0; h < 8; ++h) a[h] = p[h];
#pragma unroll
  for (int h = 0; h < 8; ++h) a[8 + h] = p[8 + h];
  return a;
}

// B: 32(K) x 16(N). lane: n = lane&15, kg = (lane<16)?0:16; halves -> k = k0+kg+h
// base points at column n=0; ldn = per-column element stride
__device__ inline v16h load_B_f16(const half_t* base, int ldn, int lane, int k0) {
  const int n  = lane & 15;
  const int kg = (lane < 16) ? 0 : 16;
  const half_t* p = base + (size_t)n * ldn + k0 + kg;
  v16h b;
#pragma unroll
  for (int h = 0; h < 16; ++h) b[h] = p[h];
  return b;
}

__device__ inline v16h load_B_f32(const float* base, int ldn, int lane, int k0) {
  const int n  = lane & 15;
  const int kg = (lane < 16) ? 0 : 16;
  const float* p = base + (size_t)n * ldn + k0 + kg;
  v16h b;
#pragma unroll
  for (int h = 0; h < 16; ++h) b[h] = (half_t)p[h];
  return b;
}

// ---------------- kernel 1: depthwise 7x7 conv + bias, dual-layout f16 -----
__global__ void dwconv7_kernel(const float* __restrict__ x,
                               const float* __restrict__ w,
                               const float* __restrict__ bias,
                               half_t* __restrict__ hw_pc,   // [NW][P][C]
                               half_t* __restrict__ hw_cp) { // [NW][C][P]
  const int idx = blockIdx.x * blockDim.x + threadIdx.x;
  const int ww = idx & (HW_DIM - 1);
  const int hh = (idx >> 7) & (HW_DIM - 1);
  const int c  = (idx >> 14) & (C_CH - 1);
  const int b  = idx >> 20;
  const float* xs = x + (((size_t)b * C_CH + c) * HW_DIM) * HW_DIM;
  const float* wc = w + c * 49;
  float acc = bias[c];
#pragma unroll
  for (int dy = -3; dy <= 3; ++dy) {
    const int yy = hh + dy;
    if (yy < 0 || yy >= HW_DIM) continue;
#pragma unroll
    for (int dx = -3; dx <= 3; ++dx) {
      const int xx = ww + dx;
      if (xx < 0 || xx >= HW_DIM) continue;
      acc += xs[yy * HW_DIM + xx] * wc[(dy + 3) * 7 + (dx + 3)];
    }
  }
  const int n = b * 16 + (hh >> 5) * 4 + (ww >> 5);
  const int p = (hh & 31) * 32 + (ww & 31);
  const half_t v = (half_t)acc;
  hw_pc[((size_t)n * P_ + p) * C_CH + c] = v;
  hw_cp[((size_t)n * C_CH + c) * P_ + p] = v;
}

// ---------------- kernel 2: qT/kT = (W @ hw)^T + b, stored [n][p][c] -------
__global__ void __launch_bounds__(256)
qk_kernel(const half_t* __restrict__ hw_pc,
          const float* __restrict__ w2, const float* __restrict__ b2,
          const float* __restrict__ w3, const float* __restrict__ b3,
          half_t* __restrict__ qT, half_t* __restrict__ kT) {
  const int lane = threadIdx.x & 31;
  const int wid  = threadIdx.x >> 5;
  const int n    = blockIdx.y;
  const int p0   = blockIdx.x * 128 + wid * 16;
  const float* W  = (blockIdx.z == 0) ? w2 : w3;
  const float* Bi = (blockIdx.z == 0) ? b2 : b3;
  half_t* dst     = (blockIdx.z == 0) ? qT : kT;

  const half_t* abase = hw_pc + ((size_t)n * P_ + p0) * C_CH;
  const v16h a0 = load_A_f16(abase, C_CH, lane, 0);   // hw^T rows p, K=c[0..31]
  const v16h a1 = load_A_f16(abase, C_CH, lane, 32);  // K=c[32..63]

  const int nn = lane & 15;
  const int mg = (lane < 16) ? 0 : 8;
  half_t* out = dst + (size_t)n * P_ * C_CH;
  for (int ot = 0; ot < 4; ++ot) {
    const v16h bf0 = load_B_f32(W + (size_t)(ot * 16) * C_CH, C_CH, lane, 0);
    const v16h bf1 = load_B_f32(W + (size_t)(ot * 16) * C_CH, C_CH, lane, 32);
    v8f acc = {};
    acc = __builtin_amdgcn_wmma_f32_16x16x32_f16(false, a0, false, bf0, (short)0, acc, false, false);
    acc = __builtin_amdgcn_wmma_f32_16x16x32_f16(false, a1, false, bf1, (short)0, acc, false, false);
    const int o = ot * 16 + nn;
    const float bv = Bi[o];
#pragma unroll
    for (int r = 0; r < 8; ++r)
      out[(size_t)(p0 + mg + r) * C_CH + o] = (half_t)(acc[r] + bv);
  }
}

// ---------------- kernel 3: fused attention + conv4 per (window, 16 queries)
__global__ void __launch_bounds__(256)
attn_kernel(const half_t* __restrict__ qT, const half_t* __restrict__ kT,
            const half_t* __restrict__ hw_cp,
            const float* __restrict__ w4, const float* __restrict__ b4,
            float* __restrict__ out) {
  extern __shared__ __align__(16) char smem[];
  float*  s_scores = (float*)smem;                                  // 16x1024 f32
  half_t* s_probs  = (half_t*)(smem + 65536);                       // 16x1024 f16
  float*  s_opart  = (float*)(smem + 65536 + 32768);                // 2x16x64 f32
  half_t* s_ot     = (half_t*)(smem + 65536 + 32768 + 8192);        // 16x64 f16
  float*  s_red    = (float*)(smem + 65536 + 32768 + 8192 + 2048);  // 16x16 f32
  float*  s_stat   = (float*)(smem + 65536 + 32768 + 8192 + 2048 + 1024); // 16
  float*  s_inv    = s_stat + 16;                                   // 16

  const int tid  = threadIdx.x;
  const int lane = tid & 31;
  const int wid  = tid >> 5;
  const int n    = blockIdx.y;
  const int j0   = blockIdx.x * 16;

  const half_t* qn  = qT + (size_t)n * P_ * C_CH;
  const half_t* kn  = kT + (size_t)n * P_ * C_CH;
  const half_t* hwn = hw_cp + (size_t)n * C_CH * P_;

  const int nn = lane & 15;
  const int mg = (lane < 16) ? 0 : 8;

  // phase 1: T[j,p] = scale * sum_c q[c,j0+j] k[c,p]
  const v16h aq0 = load_A_f16(qn + (size_t)j0 * C_CH, C_CH, lane, 0);
  const v16h aq1 = load_A_f16(qn + (size_t)j0 * C_CH, C_CH, lane, 32);
  for (int pt = wid; pt < 64; pt += 8) {
    const v16h bk0 = load_B_f16(kn + (size_t)(pt * 16) * C_CH, C_CH, lane, 0);
    const v16h bk1 = load_B_f16(kn + (size_t)(pt * 16) * C_CH, C_CH, lane, 32);
    v8f s = {};
    s = __builtin_amdgcn_wmma_f32_16x16x32_f16(false, aq0, false, bk0, (short)0, s, false, false);
    s = __builtin_amdgcn_wmma_f32_16x16x32_f16(false, aq1, false, bk1, (short)0, s, false, false);
    const int col = pt * 16 + nn;
#pragma unroll
    for (int r = 0; r < 8; ++r)
      s_scores[(mg + r) * P_ + col] = s[r] * 0.125f;  // C^-0.5 = 1/8
  }
  __syncthreads();

  // phase 2: row softmax over p (keep unnormalized exp; apply 1/sum to O)
  {
    const int row = tid >> 4, sl = tid & 15;
    float mx = -3.0e38f;
    for (int p = sl; p < P_; p += 16) mx = fmaxf(mx, s_scores[row * P_ + p]);
    s_red[row * 16 + sl] = mx;
    __syncthreads();
    if (sl == 0) {
      float m2 = s_red[row * 16];
      for (int i = 1; i < 16; ++i) m2 = fmaxf(m2, s_red[row * 16 + i]);
      s_stat[row] = m2;
    }
    __syncthreads();
    const float rm = s_stat[row];
    float sum = 0.f;
    for (int p = sl; p < P_; p += 16) {
      const float e = __expf(s_scores[row * P_ + p] - rm);
      s_probs[row * P_ + p] = (half_t)e;
      sum += e;
    }
    s_red[row * 16 + sl] = sum;
    __syncthreads();
    if (sl == 0) {
      float t = 0.f;
      for (int i = 0; i < 16; ++i) t += s_red[row * 16 + i];
      s_inv[row] = 1.0f / t;
    }
    __syncthreads();
  }

  // phase 3: O^T[j,c] = sum_p probs[j,p] * hw[c,p]   (K=1024 split over 2 halves)
  {
    const int ct = wid & 3;
    const int kh = wid >> 2;
    v8f acc = {};
    for (int s = 0; s < 16; ++s) {
      const int p0 = (kh * 16 + s) * 32;
      const v16h pa = load_A_f16(s_probs, P_, lane, p0);
      const v16h hb = load_B_f16(hwn + (size_t)(ct * 16) * P_, P_, lane, p0);
      acc = __builtin_amdgcn_wmma_f32_16x16x32_f16(false, pa, false, hb, (short)0, acc, false, false);
    }
    float* part = s_opart + kh * (16 * 64);
#pragma unroll
    for (int r = 0; r < 8; ++r)
      part[(mg + r) * 64 + ct * 16 + nn] = acc[r];
  }
  __syncthreads();
  for (int e = tid; e < 16 * 64; e += 256) {
    const int m = e >> 6;
    s_ot[e] = (half_t)((s_opart[e] + s_opart[1024 + e]) * s_inv[m]);
  }
  __syncthreads();

  // phase 4: final^T[j,o] = sum_c O^T[j,c] W4[o,c] + b4[o], scatter to NCHW
  if (wid < 4) {
    const int ot = wid;
    const v16h a0 = load_A_f16(s_ot, 64, lane, 0);
    const v16h a1 = load_A_f16(s_ot, 64, lane, 32);
    const v16h b0 = load_B_f32(w4 + (size_t)(ot * 16) * C_CH, C_CH, lane, 0);
    const v16h b1 = load_B_f32(w4 + (size_t)(ot * 16) * C_CH, C_CH, lane, 32);
    v8f d = {};
    d = __builtin_amdgcn_wmma_f32_16x16x32_f16(false, a0, false, b0, (short)0, d, false, false);
    d = __builtin_amdgcn_wmma_f32_16x16x32_f16(false, a1, false, b1, (short)0, d, false, false);
    const int o  = ot * 16 + nn;
    const float bv = b4[o];
    const int bb = n >> 4, whb = (n >> 2) & 3, wwb = n & 3;
#pragma unroll
    for (int r = 0; r < 8; ++r) {
      const int j  = j0 + mg + r;
      const int hh = whb * 32 + (j >> 5);
      const int ww = wwb * 32 + (j & 31);
      out[(((size_t)bb * C_CH + o) * HW_DIM + hh) * HW_DIM + ww] = d[r] + bv;
    }
  }
}

// ---------------- launch ---------------------------------------------------
extern "C" void kernel_launch(void* const* d_in, const int* in_sizes, int n_in,
                              void* d_out, int out_size, void* d_ws, size_t ws_size,
                              hipStream_t stream) {
  (void)in_sizes; (void)n_in; (void)out_size; (void)ws_size;
  const float* x  = (const float*)d_in[0];
  const float* w1 = (const float*)d_in[1];
  const float* b1 = (const float*)d_in[2];
  const float* w2 = (const float*)d_in[3];
  const float* b2 = (const float*)d_in[4];
  const float* w3 = (const float*)d_in[5];
  const float* b3 = (const float*)d_in[6];
  const float* w4 = (const float*)d_in[7];
  const float* b4 = (const float*)d_in[8];
  float* out = (float*)d_out;

  const size_t tile = (size_t)NW_ * P_ * C_CH;  // 4M halves = 8 MB each
  half_t* hw_pc = (half_t*)d_ws;
  half_t* hw_cp = hw_pc + tile;
  half_t* qT    = hw_cp + tile;
  half_t* kT    = qT + tile;

  dwconv7_kernel<<<(B_ * C_CH * HW_DIM * HW_DIM) / 256, 256, 0, stream>>>(
      x, w1, b1, hw_pc, hw_cp);

  qk_kernel<<<dim3(P_ / 128, NW_, 2), 256, 0, stream>>>(
      hw_pc, w2, b2, w3, b3, qT, kT);

  const size_t shmem = 65536 + 32768 + 8192 + 2048 + 1024 + 128;  // ~107.5 KB
  hipFuncSetAttribute((const void*)attn_kernel,
                      hipFuncAttributeMaxDynamicSharedMemorySize, (int)shmem);
  attn_kernel<<<dim3(P_ / 16, NW_), 256, shmem, stream>>>(
      qT, kT, hw_cp, w4, b4, out);
}
